// BilinearInfo_33646773797401
// MI455X (gfx1250) — matrix operand order repaired and checked by reading, not compiled, not run on hardware
//
#include <hip/hip_runtime.h>

// CPC forward for MI455X (gfx1250, wave32).
// Prep   : split fp32 -> (bf16 hi, bf16 lo) once: w[k] kept [o][c]; z transposed
//          to rows r=((h*16+w)*32+b) x cols c (one tensor serves all 5 k-slices).
// Stage 1: ztwk[k] = zk @ w[k]^T with 3x v_wmma_f32_16x16x32_bf16 per K=32 step
//          (Ah*Bh + Ah*Bl + Al*Bh, fp32 accum ~ fp32 accuracy at bf16 rate).
//          Tiles move via async global->LDS b128 DMA (ASYNCcnt); fragments are
//          packed-uint bitcasts, no conversion VALU in the hot loop.
// Stage 2: positive + 16 negative InfoNCE scores; exact JAX threefry2x32.
// Stage 3: zero-fill true_f region.

#define C_DIM 512
#define B_DIM 32
#define W_DIM 16
#define H_DIM 16
#define NEG   16
#define KP    5
#define NFULL (H_DIM * W_DIM * B_DIM)  // 8192 rows in transposed z

typedef __attribute__((ext_vector_type(16))) __bf16 v16bf;
typedef __attribute__((ext_vector_type(8)))  float  v8f;
typedef __attribute__((ext_vector_type(4)))  unsigned u32x4;
typedef __attribute__((ext_vector_type(4)))  int      i32x4;

typedef __attribute__((address_space(1))) int   as1_int;
typedef __attribute__((address_space(3))) int   as3_int;
typedef __attribute__((address_space(1))) i32x4 as1_i32x4;
typedef __attribute__((address_space(3))) i32x4 as3_i32x4;

#if defined(__gfx1250__) && __has_builtin(__builtin_amdgcn_s_wait_asynccnt)
#if __has_builtin(__builtin_amdgcn_global_load_async_to_lds_b128)
#define ASYNC_MODE 2
#elif __has_builtin(__builtin_amdgcn_global_load_async_to_lds_b32)
#define ASYNC_MODE 1
#else
#define ASYNC_MODE 0
#endif
#else
#define ASYNC_MODE 0
#endif

// copy 16 bytes global -> LDS
__device__ __forceinline__ void g2l_copy16(const void* g, void* l) {
#if ASYNC_MODE == 2
  __builtin_amdgcn_global_load_async_to_lds_b128(
      (as1_i32x4*)(unsigned long long)g,
      (as3_i32x4*)(unsigned)(unsigned long long)l, 0, 0);
#elif ASYNC_MODE == 1
#pragma unroll
  for (int q = 0; q < 4; ++q)
    __builtin_amdgcn_global_load_async_to_lds_b32(
        (as1_int*)((unsigned long long)g + 4ull * q),
        (as3_int*)(unsigned)((unsigned long long)l + 4u * q), 0, 0);
#else
  *(u32x4*)l = *(const u32x4*)g;
#endif
}

__device__ __forceinline__ void g2l_drain() {
#if ASYNC_MODE > 0
  __builtin_amdgcn_s_wait_asynccnt(0);
#endif
}

// assemble v16bf fragment from 8 LDS dwords (two aligned uint4 reads)
__device__ __forceinline__ v16bf frag_from(const unsigned* row, int d0, int d1) {
  union {
    unsigned u[8];
    v16bf v;
  } f;
  u32x4 x0 = *(const u32x4*)(row + d0);
  u32x4 x1 = *(const u32x4*)(row + d1);
  f.u[0] = x0[0]; f.u[1] = x0[1]; f.u[2] = x0[2]; f.u[3] = x0[3];
  f.u[4] = x1[0]; f.u[5] = x1[1]; f.u[6] = x1[2]; f.u[7] = x1[3];
  return f.v;
}

// ---- exact JAX threefry2x32 (20 rounds) ----
__device__ __forceinline__ void threefry2x32(unsigned k0, unsigned k1,
                                             unsigned x0, unsigned x1,
                                             unsigned* o0, unsigned* o1) {
  unsigned ks[3] = {k0, k1, k0 ^ k1 ^ 0x1BD11BDAu};
  const int r0[4] = {13, 15, 26, 6};
  const int r1[4] = {17, 29, 16, 24};
  x0 += ks[0];
  x1 += ks[1];
#pragma unroll
  for (int i = 0; i < 5; ++i) {
    const int* R = (i & 1) ? r1 : r0;
#pragma unroll
    for (int j = 0; j < 4; ++j) {
      x0 += x1;
      x1 = (x1 << R[j]) | (x1 >> (32 - R[j]));
      x1 ^= x0;
    }
    x0 += ks[(i + 1) % 3];
    x1 += ks[(i + 2) % 3] + (unsigned)(i + 1);
  }
  *o0 = x0;
  *o1 = x1;
}

// =====================  Prep: fp32 -> bf16 hi/lo split  =====================
__global__ __launch_bounds__(256) void cpc_prep_w(const float* __restrict__ w,
                                                  __bf16* __restrict__ whi,
                                                  __bf16* __restrict__ wlo) {
  size_t i = (size_t)blockIdx.x * 256 + threadIdx.x;  // KP*512*512 total
  float x = w[i];
  __bf16 h = (__bf16)x;
  whi[i] = h;
  wlo[i] = (__bf16)(x - (float)h);
}

__global__ __launch_bounds__(256) void cpc_prep_z(const float* __restrict__ z,
                                                  __bf16* __restrict__ zhi,
                                                  __bf16* __restrict__ zlo) {
  const int r = blockIdx.x;  // 0..8191: ((h*16+w)*32+b)
  const int b = r & 31;
  const int tt = r >> 5;
  const int wx = tt & 15;
  const int hh = tt >> 4;
  const float* zp = z + (size_t)b * (C_DIM * H_DIM * W_DIM) + hh * W_DIM + wx;
#pragma unroll
  for (int q = 0; q < 2; ++q) {
    int c = threadIdx.x + q * 256;
    float x = zp[(size_t)c * (H_DIM * W_DIM)];
    __bf16 h = (__bf16)x;
    zhi[(size_t)r * C_DIM + c] = h;
    zlo[(size_t)r * C_DIM + c] = (__bf16)(x - (float)h);
  }
}

// =====================  Stage 1: split-bf16 WMMA GEMM  =====================
// 64x64 output tile per 128-thread block (4 waves); K stepped by 32.
__global__ __launch_bounds__(128) void cpc_gemm(const __bf16* __restrict__ zhi,
                                                const __bf16* __restrict__ zlo,
                                                const __bf16* __restrict__ whi,
                                                const __bf16* __restrict__ wlo,
                                                const int* __restrict__ skip,
                                                float* __restrict__ zt) {
  // rows: 16 payload dwords (32 bf16) + 4 pad -> 80B stride, 16B-aligned rows
  __shared__ unsigned AsH[64][20];
  __shared__ unsigned AsL[64][20];
  __shared__ unsigned BsH[64][20];
  __shared__ unsigned BsL[64][20];

  const int sk = skip[0];
  const int kp = blockIdx.z;  // k = kp+1
  const int off = kp + 1 + sk;
  const int Hk = H_DIM - off;
  if (Hk <= 0) return;
  const int nrows = Hk * (W_DIM * B_DIM);
  const int m0 = blockIdx.x * 64;
  if (m0 >= nrows) return;
  const int n0 = blockIdx.y * 64;

  long base = 0;
  for (int j = 0; j < kp; ++j) {
    int Hj = H_DIM - (j + 1 + sk);
    if (Hj > 0) base += (long)Hj * 512 * 512;
  }
  float* __restrict__ ztk = zt + base;
  const __bf16* __restrict__ whik = whi + (size_t)kp * C_DIM * C_DIM;
  const __bf16* __restrict__ wlok = wlo + (size_t)kp * C_DIM * C_DIM;

  const int t = threadIdx.x;
  const int lane = t & 31;
  const int wv = t >> 5;
  const int lp = lane & 15;
  const int hi2 = lane >> 4;
  const int aoff = off * 512;  // zk row -> full-z row offset

  v8f acc[4] = {};

  for (int kk = 0; kk < C_DIM; kk += 32) {
    // stage 4 bf16 tiles (A hi/lo 64x32, B hi/lo 64x32) via async b128 DMA
#pragma unroll
    for (int e = 0; e < 2; ++e) {
      int li = t + e * 128;      // 0..255
      int row = li >> 2;         // 0..63
      int seg = li & 3;          // 16B segment within 64B row
      size_t ga = (size_t)(m0 + row + aoff) * C_DIM + kk + seg * 8;
      size_t gb = (size_t)(n0 + row) * C_DIM + kk + seg * 8;
      g2l_copy16(zhi + ga, &AsH[row][seg * 4]);
      g2l_copy16(zlo + ga, &AsL[row][seg * 4]);
      g2l_copy16(whik + gb, &BsH[row][seg * 4]);
      g2l_copy16(wlok + gb, &BsL[row][seg * 4]);
    }
    g2l_drain();
    __syncthreads();

    // A fragments: lane lp = M row; dwords [4*hi2..] and [8+4*hi2..]
    v16bf ah = frag_from(&AsH[wv * 16 + lp][0], 4 * hi2, 8 + 4 * hi2);
    v16bf al = frag_from(&AsL[wv * 16 + lp][0], 4 * hi2, 8 + 4 * hi2);
#pragma unroll
    for (int nt = 0; nt < 4; ++nt) {
      // B fragments: lane lp = N col; dwords [4*hi2..] and [8+4*hi2..]
      int col = nt * 16 + lp;
      v16bf bh = frag_from(&BsH[col][0], 4 * hi2, 8 + 4 * hi2);
      v16bf bl = frag_from(&BsL[col][0], 4 * hi2, 8 + 4 * hi2);
      acc[nt] = __builtin_amdgcn_wmma_f32_16x16x32_bf16(false, ah, false, bh,
                                                        (short)0, acc[nt], false, false);
      acc[nt] = __builtin_amdgcn_wmma_f32_16x16x32_bf16(false, ah, false, bl,
                                                        (short)0, acc[nt], false, false);
      acc[nt] = __builtin_amdgcn_wmma_f32_16x16x32_bf16(false, al, false, bh,
                                                        (short)0, acc[nt], false, false);
    }
    __syncthreads();
  }

  // D store: f32 C/D 16x16 layout (lane = N, VGPR v -> M = v + 8*(lane>=16))
#pragma unroll
  for (int nt = 0; nt < 4; ++nt) {
#pragma unroll
    for (int v = 0; v < 8; ++v) {
      int r = m0 + wv * 16 + v + 8 * hi2;
      int colg = n0 + nt * 16 + lp;
      ztk[(size_t)r * C_DIM + colg] = acc[nt][v];
    }
  }
}

// ================  Stage 2: InfoNCE scores (1 wave per row)  ================
__global__ __launch_bounds__(256) void cpc_score(const float* __restrict__ cc,
                                                 const int* __restrict__ skip,
                                                 const float* __restrict__ zt,
                                                 float* __restrict__ out) {
  const int sk = skip[0];
  const int kp = blockIdx.y;
  const int off = kp + 1 + sk;
  const int Hk = H_DIM - off;
  if (Hk <= 0) return;
  const int nrows = Hk * 512;

  long base = 0, obase = 0;
  for (int j = 0; j < kp; ++j) {
    int Hj = H_DIM - (j + 1 + sk);
    if (Hj > 0) {
      base += (long)Hj * 512 * 512;
      obase += (long)B_DIM * (NEG + 1) * Hj * W_DIM;
    }
  }
  const float* __restrict__ ztk = zt + base;

  const int lane = threadIdx.x & 31;
  const int wv = threadIdx.x >> 5;
  const int r = blockIdx.x * 8 + wv;
  if (r >= nrows) return;  // wave-uniform

  const int b = r & 31;
  const int tt = r >> 5;
  const int wx = tt & 15;
  const int h = tt >> 4;

  float ctx[16];
  const float* cp = cc + (size_t)b * (C_DIM * H_DIM * W_DIM) + h * W_DIM + wx;
#pragma unroll
  for (int j = 0; j < 16; ++j)
    ctx[j] = cp[(size_t)(lane + 32 * j) * (H_DIM * W_DIM)];

  const float* zr = ztk + (size_t)r * C_DIM;
  float p = 0.f;
#pragma unroll
  for (int j = 0; j < 16; ++j) p += ctx[j] * zr[lane + 32 * j];
#pragma unroll
  for (int d = 16; d > 0; d >>= 1) p += __shfl_xor(p, d, 32);
  if (lane == 0)
    out[obase + ((long)(b * (NEG + 1)) * Hk + h) * W_DIM + wx] = p;

  unsigned fk0, fk1;
  threefry2x32(0u, 123u, 0u, (unsigned)(kp + 1), &fk0, &fk1);
  unsigned m = (unsigned)nrows * 16u;
  unsigned hv = m >> 1;  // even m, no pad
  unsigned i = (unsigned)r * 16u + (unsigned)(lane & 15);
  unsigned jj = (i < hv) ? i : (i - hv);
  unsigned o0, o1;
  threefry2x32(fk0, fk1, jj, hv + jj, &o0, &o1);
  unsigned bits = (i < hv) ? o0 : o1;
  int myidx = (int)(bits % (unsigned)nrows);

  for (int s = 0; s < NEG; ++s) {
    int rs = __shfl(myidx, s, 32);
    const float* zs = ztk + (size_t)rs * C_DIM;  // coalesced 2KB row, L2-hot
    float q = 0.f;
#pragma unroll
    for (int j = 0; j < 16; ++j) q += ctx[j] * zs[lane + 32 * j];
#pragma unroll
    for (int d = 16; d > 0; d >>= 1) q += __shfl_xor(q, d, 32);
    if (lane == 0)
      out[obase + ((long)(b * (NEG + 1) + (s + 1)) * Hk + h) * W_DIM + wx] = q;
  }
}

// ===================  Stage 3: zero-fill true_f region  ===================
__global__ void cpc_zero(float* __restrict__ out, const int* __restrict__ skip,
                         int out_size) {
  const int sk = skip[0];
  long logf = 0;
  for (int j = 0; j < KP; ++j) {
    int Hj = H_DIM - (j + 1 + sk);
    if (Hj > 0) logf += (long)B_DIM * (NEG + 1) * Hj * W_DIM;
  }
  long e = (long)blockIdx.x * blockDim.x + threadIdx.x;
  if (e >= logf && e < out_size) out[e] = 0.0f;
}

extern "C" void kernel_launch(void* const* d_in, const int* in_sizes, int n_in,
                              void* d_out, int out_size, void* d_ws, size_t ws_size,
                              hipStream_t stream) {
  const float* z = (const float*)d_in[0];
  const float* c = (const float*)d_in[1];
  const float* w = (const float*)d_in[2];
  const int* skip = (const int*)d_in[3];
  float* out = (float*)d_out;

  // out_size = 9216 * sum(Hk); sum(Hk) = 65 - 5*skip -> recover skip on host
  int sumH = out_size / (B_DIM * W_DIM * (NEG + 2));
  int sk = (65 - sumH) / 5;
  if (sk < 0) sk = 0;
  if (sk > 10) sk = 10;
  int Hmax = 15 - sk;
  if (Hmax < 1) Hmax = 1;

  // workspace layout: [ztwk fp32 | zhi | zlo | whi | wlo]
  char* wsb = (char*)d_ws;
  float* ztw = (float*)wsb;
  size_t ztf = (size_t)sumH * C_DIM * C_DIM;  // floats
  __bf16* zhi = (__bf16*)(wsb + ztf * 4);
  __bf16* zlo = zhi + (size_t)NFULL * C_DIM;
  __bf16* whi = zlo + (size_t)NFULL * C_DIM;
  __bf16* wlo = whi + (size_t)KP * C_DIM * C_DIM;

  cpc_prep_w<<<(KP * C_DIM * C_DIM) / 256, 256, 0, stream>>>(w, whi, wlo);
  cpc_prep_z<<<NFULL, 256, 0, stream>>>(z, zhi, zlo);

  dim3 gg((unsigned)(Hmax * 8), 8, KP);  // M-tiles x N-tiles x k
  cpc_gemm<<<gg, 128, 0, stream>>>(zhi, zlo, whi, wlo, skip, ztw);

  dim3 gs((unsigned)(Hmax * 64), KP, 1);  // 8 rows/block x k
  cpc_score<<<gs, 256, 0, stream>>>(c, skip, ztw, out);

  int zb = (out_size + 255) / 256;
  cpc_zero<<<zb, 256, 0, stream>>>(out, skip, out_size);
}
